// RingAttentionLlama_16346645529050
// MI455X (gfx1250) — matrix operand order repaired and checked by reading, not compiled, not run on hardware
//
#include <hip/hip_runtime.h>
#include <cstdint>
#include <cstddef>

// ---------------------------------------------------------------------------
// Types for CDNA5 WMMA (wave32)
// ---------------------------------------------------------------------------
typedef __bf16 bf16;
typedef __attribute__((ext_vector_type(16))) __bf16 v16bf;
typedef __attribute__((ext_vector_type(8)))  __bf16 v8bf;
typedef __attribute__((ext_vector_type(8)))  float  v8f;
typedef __attribute__((ext_vector_type(4)))  unsigned int u32x4;
typedef __attribute__((ext_vector_type(8)))  unsigned int u32x8;

constexpr int   kB    = 2;
constexpr int   kS    = 2048;
constexpr int   kD    = 4096;
constexpr int   kNH   = 32;
constexpr int   kNKV  = 8;
constexpr int   kHD   = 128;
constexpr int   kNREP = 4;
constexpr float kScale = 0.08838834764831845f;   // 1/sqrt(128)

__device__ __forceinline__ v16bf bfcat(v8bf lo, v8bf hi) {
  return __builtin_shufflevector(lo, hi, 0,1,2,3,4,5,6,7,8,9,10,11,12,13,14,15);
}
__device__ __forceinline__ v8f vzero8() {
  v8f z = {0.f,0.f,0.f,0.f,0.f,0.f,0.f,0.f};
  return z;
}
__device__ __forceinline__ v8f wmma_bf16(v16bf a, v16bf b, v8f c) {
  // D = A(16x32) * B(32x16) + C(16x16), fp32 accumulate
  return __builtin_amdgcn_wmma_f32_16x16x32_bf16(false, a, false, b, (short)0, c,
                                                 false, false);
}

// ---------------------------------------------------------------------------
// Tensor Data Mover: 2D tile load Global->LDS (bf16 elements, data_size=2B).
// D# per CDNA5 ISA. Issued by one wave; tracked by
// TENSORcnt. pad_* encodings: interval 2^(n+1) DWORDs, amount (n+1) DWORDs.
// ---------------------------------------------------------------------------
__device__ __forceinline__ void tdm_load_2d_bf16(
    unsigned int lds_addr, const void* gptr,
    unsigned int tensor_d0, unsigned int tensor_d1,
    unsigned int tile_d0, unsigned int tile_d1,
    unsigned long long stride_d0_elems,
    unsigned int pad_interval_enc, unsigned int pad_amount_enc,
    bool pad_en) {
  unsigned long long ga = (unsigned long long)(uintptr_t)gptr;
  u32x4 g0;
  g0[0] = 1u;                                            // count=1, user D#
  g0[1] = lds_addr;                                      // LDS byte address
  g0[2] = (unsigned int)(ga & 0xFFFFFFFFu);              // global_addr[31:0]
  g0[3] = (unsigned int)((ga >> 32) & 0x01FFFFFFu) | (2u << 30);  // type=2
  u32x8 g1;
  g1[0] = (1u << 16)                                     // data_size = 2B
        | (pad_en ? (1u << 20) : 0u)                     // pad_enable
        | (pad_interval_enc << 22)
        | (pad_amount_enc << 25);
  g1[1] = (tensor_d0 & 0xFFFFu) << 16;                   // dim0[15:0] @ [63:48]
  g1[2] = (tensor_d0 >> 16) | ((tensor_d1 & 0xFFFFu) << 16);
  g1[3] = (tensor_d1 >> 16) | (tile_d0 << 16);           // tile_dim0 @ [127:112]
  g1[4] = (tile_d1 & 0xFFFFu);                           // tile_dim1; tile_dim2=0
  g1[5] = (unsigned int)(stride_d0_elems & 0xFFFFFFFFu); // dim0_stride[31:0]
  g1[6] = (unsigned int)((stride_d0_elems >> 32) & 0xFFFFu);
  g1[7] = 0u;
  asm volatile("tensor_load_to_lds %0, %1" :: "s"(g0), "s"(g1) : "memory");
}

// ---------------------------------------------------------------------------
// fp32 -> bf16 conversion (4 elements per thread)
// ---------------------------------------------------------------------------
__global__ void cvt_f32_bf16(const float* __restrict__ s, bf16* __restrict__ d, int n) {
  int i = (blockIdx.x * blockDim.x + threadIdx.x) * 4;
  if (i + 3 < n) {
    float4 v = *(const float4*)(s + i);
    d[i + 0] = (bf16)v.x;
    d[i + 1] = (bf16)v.y;
    d[i + 2] = (bf16)v.z;
    d[i + 3] = (bf16)v.w;
  } else {
    for (int k = i; k < n; ++k) d[k] = (bf16)s[k];
  }
}

// ---------------------------------------------------------------------------
// Tiled bf16 GEMM: C(MxN) = A(MxK, row-major bf16) * B(KxN, row-major bf16)
// Block tile 128x128, K-step 32. 256 threads = 8 waves, wave grid 2x4,
// each wave: 4 M-tiles x 2 N-tiles of 16x16 -> 8 WMMA per K-step.
// A-tile is staged by the Tensor Data Mover (DMA pads rows 32 -> 40 elems).
// ---------------------------------------------------------------------------
template <bool F32OUT>
__global__ __launch_bounds__(256) void gemm_bf16(const bf16* __restrict__ A,
                                                 const bf16* __restrict__ B,
                                                 void* __restrict__ Cv,
                                                 int M, int N, int K) {
  constexpr int LDA = 40;   // 32 + 8 pad (80B rows: 16B aligned, bank-spread)
  constexpr int LDB = 40;
  __shared__ bf16 As[128 * LDA];   // [row][k]   (TDM-filled, padded)
  __shared__ bf16 Bs[128 * LDB];   // transposed: [n][k]

  const int tid  = threadIdx.x;
  const int lane = tid & 31;
  const int w    = tid >> 5;       // wave 0..7
  const int wr   = w >> 2;         // 0..1  -> rows wr*64
  const int wc   = w & 3;          // 0..3  -> cols wc*32
  const int hi16 = lane >> 4;      // lane half
  const int l16  = lane & 15;

  const int mb = blockIdx.y * 128;
  const int nb = blockIdx.x * 128;

  const unsigned int asLds = (unsigned int)(uintptr_t)(void*)As;

  v8f acc[4][2];
#pragma unroll
  for (int i = 0; i < 4; ++i)
#pragma unroll
    for (int j = 0; j < 2; ++j) acc[i][j] = vzero8();

  const int bk  = tid >> 3;          // B-tile k-row this thread loads
  const int bn0 = (tid & 7) * 16;    // B-tile n base

  for (int kt = 0; kt < K; kt += 32) {
    __syncthreads();
    // ---- TDM: stage A tile (128 rows x 32 cols), one DMA for the block ----
    if (w == 0) {
      // pad_interval enc 3 = 16 DWORDs (64B = one 32-elem row),
      // pad_amount   enc 3 = 4 DWORDs (8 elems) -> LDS row stride 40.
      tdm_load_2d_bf16(asLds, A + (size_t)mb * K + kt,
                       (unsigned)K, (unsigned)M, /*tile*/32u, 128u,
                       (unsigned long long)K, 3u, 3u, true);
    }
    // ---- stage B tile (32 x 128) transposed -> Bs[n][k] ----
    {
      const bf16* gp = B + (size_t)(kt + bk) * N + nb + bn0;
      v8bf x0 = *(const v8bf*)gp;
      v8bf x1 = *(const v8bf*)(gp + 8);
#pragma unroll
      for (int i = 0; i < 8; ++i) {
        Bs[(bn0 + i)     * LDB + bk] = x0[i];
        Bs[(bn0 + 8 + i) * LDB + bk] = x1[i];
      }
      if (kt + 32 < K) __builtin_prefetch(gp + (size_t)32 * N, 0, 1);
    }
    if (w == 0) __builtin_amdgcn_s_wait_tensorcnt(0);
    __syncthreads();

    // ---- B fragments (lane = column, K split 0-7/8-15 then +16) ----
    v16bf bfrag[2];
#pragma unroll
    for (int j = 0; j < 2; ++j) {
      const bf16* p = Bs + (wc * 32 + j * 16 + l16) * LDB + hi16 * 8;
      bfrag[j] = bfcat(*(const v8bf*)p, *(const v8bf*)(p + 16));
    }
    // ---- A fragments (lane = row, K split 0-7/8-15 then +16) + WMMA ----
#pragma unroll
    for (int i = 0; i < 4; ++i) {
      const bf16* p = As + (wr * 64 + i * 16 + l16) * LDA + hi16 * 8;
      v16bf afrag = bfcat(*(const v8bf*)p, *(const v8bf*)(p + 16));
#pragma unroll
      for (int j = 0; j < 2; ++j) acc[i][j] = wmma_bf16(afrag, bfrag[j], acc[i][j]);
    }
  }

  // ---- write-out: C layout: VGPR r -> row r + hi16*8, lane -> col l16 ----
#pragma unroll
  for (int i = 0; i < 4; ++i) {
    const int row0 = mb + wr * 64 + i * 16 + hi16 * 8;
#pragma unroll
    for (int j = 0; j < 2; ++j) {
      const int col = nb + wc * 32 + j * 16 + l16;
#pragma unroll
      for (int r = 0; r < 8; ++r) {
        const size_t off = (size_t)(row0 + r) * N + col;
        if (F32OUT) ((float*)Cv)[off] = acc[i][j][r];
        else        ((bf16*)Cv)[off]  = (bf16)acc[i][j][r];
      }
    }
  }
}

// ---------------------------------------------------------------------------
// Causal flash attention (GQA). Grid: x = b*NH + h (64), y = q-tile (S/128).
// Block = 256 threads (8 waves); wave w owns query rows [w*16, w*16+15].
// Streams 64-key tiles; K-tile staged by TDM, online softmax in WMMA C-layout.
// ---------------------------------------------------------------------------
__global__ __launch_bounds__(256) void flash_attn(const bf16* __restrict__ Q,
                                                  const bf16* __restrict__ K,
                                                  const bf16* __restrict__ V,
                                                  bf16* __restrict__ O) {
  constexpr int LDK = 136;  // 128 + 8 pad (TDM pad: 64-DW interval, 4-DW pad)
  constexpr int LDV = 72;   // 64 + 8 pad
  constexpr int LDP = 72;
  __shared__ bf16 Ks[64  * LDK];  // [key][hd]  (TDM-filled, padded)
  __shared__ bf16 Vt[128 * LDV];  // [hd][key]  (transposed V)
  __shared__ bf16 Ps[128 * LDP];  // [qrow][key], wave-private 16-row strips

  const int tid  = threadIdx.x;
  const int lane = tid & 31;
  const int w    = tid >> 5;
  const int hi16 = lane >> 4;
  const int l16  = lane & 15;

  const int b  = blockIdx.x / kNH;
  const int h  = blockIdx.x % kNH;
  const int kv = h / kNREP;
  const int q0 = blockIdx.y * 128;

  const unsigned int ksLds = (unsigned int)(uintptr_t)(void*)Ks;

  // ---- load this lane's Q A-fragments (row = q0 + w*16 + l16) ----
  const int qrow = q0 + w * 16 + l16;
  const bf16* qbase = Q + (((size_t)b * kS + qrow) * kNH + h) * kHD;
  v16bf qf[4];
#pragma unroll
  for (int c = 0; c < 4; ++c) {
    const bf16* p = qbase + c * 32 + hi16 * 8;
    qf[c] = bfcat(*(const v8bf*)p, *(const v8bf*)(p + 16));
  }

  float m_i[8], s_i[8];
  v8f oacc[8];
#pragma unroll
  for (int r = 0; r < 8; ++r) { m_i[r] = -3.0e38f; s_i[r] = 0.f; }
#pragma unroll
  for (int t = 0; t < 8; ++t) oacc[t] = vzero8();

  const int ldr = tid >> 2;          // 0..63 key row for V staging
  const int ldc = (tid & 3) * 32;    // hd col base
  const int ktiles = (q0 + 128) / 64;

  for (int kt = 0; kt < ktiles; ++kt) {
    const int kbase = kt * 64;
    __syncthreads();   // protect Ks/Vt/Ps reuse from previous iteration
    // ---- TDM: stage K tile (64 keys x 128 hd) -> Ks, rows padded to 136 ----
    if (w == 0) {
      const bf16* gp = K + (((size_t)b * kS + kbase) * kNKV + kv) * kHD;
      // pad_interval enc 5 = 64 DWORDs (256B = one 128-elem row),
      // pad_amount   enc 3 = 4 DWORDs (8 elems) -> LDS row stride 136.
      tdm_load_2d_bf16(ksLds, gp, 128u, 0x40000000u, /*tile*/128u, 64u,
                       (unsigned long long)(kNKV * kHD), 5u, 3u, true);
    }
    // ---- stage V tile transposed -> Vt[hd][key] ----
    {
      const bf16* gp = V + (((size_t)b * kS + kbase + ldr) * kNKV + kv) * kHD + ldc;
      v8bf x[4];
#pragma unroll
      for (int i = 0; i < 4; ++i) x[i] = *(const v8bf*)(gp + 8 * i);
#pragma unroll
      for (int i = 0; i < 32; ++i) Vt[(ldc + i) * LDV + ldr] = x[i >> 3][i & 7];
    }
    if (w == 0) __builtin_amdgcn_s_wait_tensorcnt(0);
    __syncthreads();

    // ---- S = (Q @ K^T) * scale, 16 rows x 64 keys per wave ----
    v8f sacc[4];
#pragma unroll
    for (int j = 0; j < 4; ++j) sacc[j] = vzero8();
#pragma unroll
    for (int c = 0; c < 4; ++c) {       // hd chunks of 32
#pragma unroll
      for (int j = 0; j < 4; ++j) {     // 16-key column tiles
        const bf16* p = Ks + (j * 16 + l16) * LDK + c * 32 + hi16 * 8;
        v16bf kf = bfcat(*(const v8bf*)p, *(const v8bf*)(p + 16));
        sacc[j] = wmma_bf16(qf[c], kf, sacc[j]);
      }
    }

    // ---- causal mask + scale (C layout: row = r + hi16*8, col = l16) ----
#pragma unroll
    for (int j = 0; j < 4; ++j) {
      const int col_g = kbase + j * 16 + l16;
#pragma unroll
      for (int r = 0; r < 8; ++r) {
        const int row_g = q0 + w * 16 + r + hi16 * 8;
        float v = sacc[j][r] * kScale;
        sacc[j][r] = (col_g > row_g) ? -3.0e38f : v;
      }
    }

    // ---- online softmax per row (reduce over 16-lane half-wave) ----
#pragma unroll
    for (int r = 0; r < 8; ++r) {
      float t = fmaxf(fmaxf(sacc[0][r], sacc[1][r]), fmaxf(sacc[2][r], sacc[3][r]));
      t = fmaxf(t, __shfl_xor(t, 1, 32));
      t = fmaxf(t, __shfl_xor(t, 2, 32));
      t = fmaxf(t, __shfl_xor(t, 4, 32));
      t = fmaxf(t, __shfl_xor(t, 8, 32));
      const float mnew = fmaxf(m_i[r], t);
      const float corr = __expf(m_i[r] - mnew);
      m_i[r] = mnew;
      float rs = 0.f;
#pragma unroll
      for (int j = 0; j < 4; ++j) {
        float e = __expf(sacc[j][r] - mnew);
        sacc[j][r] = e;
        rs += e;
      }
      rs += __shfl_xor(rs, 1, 32);
      rs += __shfl_xor(rs, 2, 32);
      rs += __shfl_xor(rs, 4, 32);
      rs += __shfl_xor(rs, 8, 32);
      s_i[r] = s_i[r] * corr + rs;
#pragma unroll
      for (int t8 = 0; t8 < 8; ++t8) oacc[t8][r] *= corr;
    }

    // ---- spill P (bf16) to wave-private LDS strip for A-layout reload ----
#pragma unroll
    for (int j = 0; j < 4; ++j)
#pragma unroll
      for (int r = 0; r < 8; ++r)
        Ps[(w * 16 + r + hi16 * 8) * LDP + j * 16 + l16] = (bf16)sacc[j][r];
    __syncthreads();

    // ---- O += P @ V  (keys in chunks of 32) ----
#pragma unroll
    for (int kc = 0; kc < 2; ++kc) {
      const bf16* pp = Ps + (w * 16 + l16) * LDP + kc * 32 + hi16 * 8;
      v16bf pf = bfcat(*(const v8bf*)pp, *(const v8bf*)(pp + 16));
#pragma unroll
      for (int t = 0; t < 8; ++t) {
        const bf16* vp = Vt + (t * 16 + l16) * LDV + kc * 32 + hi16 * 8;
        v16bf vf = bfcat(*(const v8bf*)vp, *(const v8bf*)(vp + 16));
        oacc[t] = wmma_bf16(pf, vf, oacc[t]);
      }
    }
  }

  // ---- normalize and store O (b, s, h, d) bf16 ----
  float inv[8];
#pragma unroll
  for (int r = 0; r < 8; ++r) inv[r] = 1.0f / s_i[r];
#pragma unroll
  for (int t = 0; t < 8; ++t)
#pragma unroll
    for (int r = 0; r < 8; ++r) {
      const int row = q0 + w * 16 + r + hi16 * 8;
      O[(((size_t)b * kS + row) * kNH + h) * kHD + t * 16 + l16] =
          (bf16)(oacc[t][r] * inv[r]);
    }
}

// ---------------------------------------------------------------------------
// Host launcher
// ---------------------------------------------------------------------------
extern "C" void kernel_launch(void* const* d_in, const int* in_sizes, int n_in,
                              void* d_out, int out_size, void* d_ws, size_t ws_size,
                              hipStream_t stream) {
  (void)in_sizes; (void)n_in; (void)out_size; (void)ws_size;
  const float* x  = (const float*)d_in[0];
  const float* wq = (const float*)d_in[1];
  const float* wk = (const float*)d_in[2];
  const float* wv = (const float*)d_in[3];
  const float* wo = (const float*)d_in[4];

  const size_t nBS = (size_t)kB * kS;          // 4096 rows
  const size_t nX  = nBS * kD;                 // 16,777,216
  const size_t nWq = (size_t)kD * (kNH * kHD); // 16,777,216
  const size_t nWk = (size_t)kD * (kNKV * kHD);// 4,194,304
  const size_t nWo = nWq;

  // workspace layout (bf16), ~201 MB total, all offsets 256B-aligned
  bf16* p   = (bf16*)d_ws;
  bf16* xb  = p; p += nX;
  bf16* wqb = p; p += nWq;
  bf16* wkb = p; p += nWk;
  bf16* wvb = p; p += nWk;
  bf16* wob = p; p += nWo;
  bf16* Qb  = p; p += nX;        // (b,s,h,d)
  bf16* Kb  = p; p += nBS * (kNKV * kHD);
  bf16* Vb  = p; p += nBS * (kNKV * kHD);
  bf16* Ob  = p; p += nX;        // attention output (b,s,h*d)

  auto cvt = [&](const float* s, bf16* d, size_t n) {
    cvt_f32_bf16<<<(unsigned)((n / 4 + 255) / 256), 256, 0, stream>>>(s, d, (int)n);
  };
  cvt(x,  xb,  nX);
  cvt(wq, wqb, nWq);
  cvt(wk, wkb, nWk);
  cvt(wv, wvb, nWk);
  cvt(wo, wob, nWo);

  // projections: Q (4096x4096x4096), K/V (4096x1024x4096)
  gemm_bf16<false><<<dim3(32, 32), 256, 0, stream>>>(xb, wqb, Qb, 4096, 4096, 4096);
  gemm_bf16<false><<<dim3( 8, 32), 256, 0, stream>>>(xb, wkb, Kb, 4096, 1024, 4096);
  gemm_bf16<false><<<dim3( 8, 32), 256, 0, stream>>>(xb, wvb, Vb, 4096, 1024, 4096);

  // causal flash attention: 64 (b,h) x 16 query tiles
  flash_attn<<<dim3(kB * kNH, kS / 128), 256, 0, stream>>>(Qb, Kb, Vb, Ob);

  // output projection -> fp32 d_out
  gemm_bf16<true><<<dim3(32, 32), 256, 0, stream>>>(Ob, wob, (float*)d_out,
                                                    4096, 4096, 4096);
}
